// _PointnetSAModuleBase_5755256177431
// MI455X (gfx1250) — matrix operand order repaired and checked by reading, not compiled, not run on hardware
//
#include <hip/hip_runtime.h>
#include <hip/hip_bf16.h>

typedef __bf16 v16bf __attribute__((ext_vector_type(16)));
typedef float  v8f   __attribute__((ext_vector_type(8)));

#define BATCH    8
#define NPTS     8192
#define NPOINT   2048
#define NSAMPLE  32
#define CIN      67      // 3 rel + 64 features
#define H1DIM    64
#define H2DIM    128
#define R2       0.01f   // RADIUS^2
#define KPAD     96      // CIN padded to 3 * 32

// ---------------------------------------------------------------------------
// Kernel A: ball query (+ new_xyz copy).  One block = 256 queries of one batch.
// xyz tiled through LDS (2048 pts = 24KB/tile).  Wave32 ballot prefix-scatter
// keeps indices in ascending order exactly like the reference argsort trick.
// ---------------------------------------------------------------------------
#define TILE_PTS 2048

__global__ __launch_bounds__(256)
void ball_query_kernel(const float* __restrict__ xyz,
                       float* __restrict__ out_newxyz,
                       int* __restrict__ ws_idx) {
  __shared__ float sx[TILE_PTS], sy[TILE_PTS], sz[TILE_PTS];
  __shared__ int s_cnt[8][32];
  __shared__ int s_first[8][32];

  const int b    = blockIdx.x >> 3;          // 8 blocks per batch
  const int s0   = (blockIdx.x & 7) * 256;   // query base
  const int tid  = threadIdx.x;
  const int wave = tid >> 5;
  const int lane = tid & 31;
  const float* xb = xyz + (size_t)b * NPTS * 3;

  // new_xyz slice for this block: contiguous 768 floats
  for (int i = tid; i < 256 * 3; i += 256)
    out_newxyz[((size_t)b * NPOINT + s0) * 3 + i] = xb[(size_t)s0 * 3 + i];

  s_cnt[wave][lane]   = 0;
  s_first[wave][lane] = 0;
  __syncthreads();

  for (int t = 0; t < NPTS / TILE_PTS; ++t) {
    for (int i = tid; i < TILE_PTS; i += 256) {
      int p = t * TILE_PTS + i;
      sx[i] = xb[p * 3 + 0];
      sy[i] = xb[p * 3 + 1];
      sz[i] = xb[p * 3 + 2];
    }
    __syncthreads();

    for (int qi = 0; qi < 32; ++qi) {
      int cnt = s_cnt[wave][qi];
      if (cnt >= NSAMPLE) continue;                  // wave-uniform
      const int s = s0 + wave * 32 + qi;
      const float qx = xb[s * 3 + 0], qy = xb[s * 3 + 1], qz = xb[s * 3 + 2];
      int first = s_first[wave][qi];
      int* dst = ws_idx + ((size_t)b * NPOINT + s) * NSAMPLE;

      for (int base = 0; base < TILE_PTS; base += 32) {
        const int li = base + lane;
        const float dx = qx - sx[li], dy = qy - sy[li], dz = qz - sz[li];
        const bool inb = (dx * dx + dy * dy + dz * dz) < R2;
        const unsigned m = (unsigned)__ballot(inb);    // wave32: low 32 bits
        if (m) {
          if (cnt == 0) first = t * TILE_PTS + base + __builtin_ctz(m);
          const int pos = cnt + __popc(m & ((1u << lane) - 1u));
          if (inb && pos < NSAMPLE) dst[pos] = t * TILE_PTS + li;
          cnt += __popc(m);
          if (cnt >= NSAMPLE) break;                   // uniform
        }
      }
      if (lane == 0) { s_cnt[wave][qi] = cnt; s_first[wave][qi] = first; }
    }
    __syncthreads();
  }

  // padding pass: slots >= cnt get the first in-radius index (ref semantics)
  for (int qi = 0; qi < 32; ++qi) {
    const int s = s0 + wave * 32 + qi;
    const int cnt = s_cnt[wave][qi];
    const int first = (cnt > 0) ? s_first[wave][qi] : 0;
    int* dst = ws_idx + ((size_t)b * NPOINT + s) * NSAMPLE;
    if (lane >= cnt) dst[lane] = first;
  }
}

// ---------------------------------------------------------------------------
// Kernel B: fused gather + 2-layer MLP (bf16 WMMA, f32 accum) + max-pool.
// One wave per group; weights staged in LDS in fragment-friendly bf16 layout;
// H1 bounced through per-wave LDS (col-major) to re-stripe C/D layout into
// the B-operand layout for layer 2.
// ---------------------------------------------------------------------------
#define WAVES_B 8
#define GPW     8   // groups per wave; 256 blocks * 8 waves * 8 = 16384 groups

__device__ __forceinline__ v8f wmma_bf16(v16bf a, v16bf b, v8f c) {
  return __builtin_amdgcn_wmma_f32_16x16x32_bf16(false, a, false, b,
                                                 (short)0, c, false, false);
}

__global__ __launch_bounds__(256)
void grouped_mlp_kernel(const float* __restrict__ xyz,
                        const float* __restrict__ feat,
                        const float* __restrict__ W1,
                        const float* __restrict__ b1,
                        const float* __restrict__ W2,
                        const float* __restrict__ b2,
                        const int* __restrict__ ws_idx,
                        float* __restrict__ out_feat) {
  __shared__ __align__(16) __bf16 sW1[H1DIM * KPAD];        // 12 KB
  __shared__ __align__(16) __bf16 sW2[H2DIM * H1DIM];       // 16 KB
  __shared__ __align__(16) __bf16 sH1[WAVES_B][NSAMPLE * H1DIM]; // 32 KB
  __shared__ float sb1[H1DIM];
  __shared__ float sb2[H2DIM];

  const int tid  = threadIdx.x;
  const int wave = tid >> 5;
  const int lane = tid & 31;
  const bool hi  = lane >= 16;
  const int  l16 = lane & 15;

  // stage weights/biases as bf16/f32 in LDS
  for (int i = tid; i < H1DIM * KPAD; i += 256) {
    const int m = i / KPAD, k = i % KPAD;
    sW1[i] = (__bf16)((k < CIN) ? W1[m * CIN + k] : 0.0f);
  }
  for (int i = tid; i < H2DIM * H1DIM; i += 256) sW2[i] = (__bf16)W2[i];
  if (tid < H1DIM) sb1[tid] = b1[tid];
  if (tid < H2DIM) sb2[tid] = b2[tid];
  __syncthreads();

  for (int gi = 0; gi < GPW; ++gi) {
    const int g = (blockIdx.x * WAVES_B + wave) * GPW + gi;
    const int b = g >> 11;           // / NPOINT
    const int s = g & 2047;          // % NPOINT
    const float* fb = feat + (size_t)b * H1DIM * NPTS;  // 64 x 8192
    const float* xb = xyz + (size_t)b * NPTS * 3;

    const int idx_l = ws_idx[(size_t)g * NSAMPLE + lane];
    const float qx = xb[s * 3 + 0], qy = xb[s * 3 + 1], qz = xb[s * 3 + 2];

    // per-lane sample column per ntile: gathered index + rel coords
    int   pidx[2];
    float rx[2], ry[2], rz[2];
#pragma unroll
    for (int nt = 0; nt < 2; ++nt) {
      const int col = nt * 16 + l16;
      pidx[nt] = __shfl(idx_l, col, 32);
      rx[nt] = xb[pidx[nt] * 3 + 0] - qx;
      ry[nt] = xb[pidx[nt] * 3 + 1] - qy;
      rz[nt] = xb[pidx[nt] * 3 + 2] - qz;
    }

    // ------------------- layer 1: 64x32 = W1(64x96) * G(96x32) -------------
    v8f acc1[4][2];
#pragma unroll
    for (int mt = 0; mt < 4; ++mt) { acc1[mt][0] = (v8f)0.0f; acc1[mt][1] = (v8f)0.0f; }

#pragma unroll
    for (int kc = 0; kc < 3; ++kc) {
      // build gathered B-fragments (K x N layout: elems 0..7 -> k 8*hi+0..7,
      // elems 8..15 -> k 16+8*hi+0..7)
      v16bf bfr[2];
      const int kbase = kc * 32 + (hi ? 8 : 0);
#pragma unroll
      for (int nt = 0; nt < 2; ++nt) {
        union { v16bf v; __bf16 h[16]; } u;
#pragma unroll
        for (int j = 0; j < 16; ++j) {
          const int k = kbase + j + (j >= 8 ? 8 : 0);
          float val;
          if (k == 0)      val = rx[nt];
          else if (k == 1) val = ry[nt];
          else if (k == 2) val = rz[nt];
          else {
            const int c = k - 3;
            val = (c < 64) ? fb[(size_t)c * NPTS + pidx[nt]] : 0.0f;
          }
          u.h[j] = (__bf16)val;
        }
        bfr[nt] = u.v;
      }
#pragma unroll
      for (int mt = 0; mt < 4; ++mt) {
        // A-frag: row m, elems 0..7 -> k0..k0+7, elems 8..15 -> k0+16..k0+23
        const int m  = mt * 16 + l16;
        const int k0 = kc * 32 + (hi ? 8 : 0);
        union { v16bf v; uint4 q[2]; } a;
        a.q[0] = *(const uint4*)&sW1[m * KPAD + k0];
        a.q[1] = *(const uint4*)&sW1[m * KPAD + k0 + 16];
        acc1[mt][0] = wmma_bf16(a.v, bfr[0], acc1[mt][0]);
        acc1[mt][1] = wmma_bf16(a.v, bfr[1], acc1[mt][1]);
      }
    }

    // bias + relu, re-stripe H1 through LDS (col-major [col][64] bf16)
#pragma unroll
    for (int mt = 0; mt < 4; ++mt) {
#pragma unroll
      for (int nt = 0; nt < 2; ++nt) {
        const int col = nt * 16 + l16;
        const int k0  = mt * 16 + (hi ? 8 : 0);   // C/D layout: M = r + 8*hi
        union { __bf16 h[8]; uint4 q; } u;
#pragma unroll
        for (int r = 0; r < 8; ++r)
          u.h[r] = (__bf16)fmaxf(acc1[mt][nt][r] + sb1[k0 + r], 0.0f);
        *(uint4*)&sH1[wave][col * H1DIM + k0] = u.q;
      }
    }

    // ------------------- layer 2: 128x32 = W2(128x64) * H1(64x32) ----------
#pragma unroll
    for (int mt = 0; mt < 8; ++mt) {
      v8f acc2[2] = { (v8f)0.0f, (v8f)0.0f };
#pragma unroll
      for (int kc = 0; kc < 2; ++kc) {
        const int m  = mt * 16 + l16;
        const int k0 = kc * 32 + (hi ? 8 : 0);
        union { v16bf v; uint4 q[2]; } a;
        a.q[0] = *(const uint4*)&sW2[m * H1DIM + k0];
        a.q[1] = *(const uint4*)&sW2[m * H1DIM + k0 + 16];
        const int kb = kc * 32 + (hi ? 8 : 0);
#pragma unroll
        for (int nt = 0; nt < 2; ++nt) {
          const int col = nt * 16 + l16;
          union { v16bf v; uint4 q[2]; } bb;
          bb.q[0] = *(const uint4*)&sH1[wave][col * H1DIM + kb];
          bb.q[1] = *(const uint4*)&sH1[wave][col * H1DIM + kb + 16];
          acc2[nt] = wmma_bf16(a.v, bb.v, acc2[nt]);
        }
      }
      // max over 32 samples: elementwise over ntiles, then across 16 lanes
      // of each half (xor 1,2,4,8 stays within the half on wave32)
#pragma unroll
      for (int r = 0; r < 8; ++r) {
        float v = fmaxf(acc2[0][r], acc2[1][r]);
        v = fmaxf(v, __shfl_xor(v, 1, 32));
        v = fmaxf(v, __shfl_xor(v, 2, 32));
        v = fmaxf(v, __shfl_xor(v, 4, 32));
        v = fmaxf(v, __shfl_xor(v, 8, 32));
        if (l16 == 0) {
          const int m = mt * 16 + (hi ? 8 : 0) + r;
          out_feat[((size_t)b * H2DIM + m) * NPOINT + s] = fmaxf(v + sb2[m], 0.0f);
        }
      }
    }
  }
}

// ---------------------------------------------------------------------------
extern "C" void kernel_launch(void* const* d_in, const int* in_sizes, int n_in,
                              void* d_out, int out_size, void* d_ws, size_t ws_size,
                              hipStream_t stream) {
  const float* xyz  = (const float*)d_in[0];
  const float* feat = (const float*)d_in[1];
  const float* W1   = (const float*)d_in[2];
  const float* b1   = (const float*)d_in[3];
  const float* W2   = (const float*)d_in[4];
  const float* b2   = (const float*)d_in[5];
  // d_in[6] = npoint (2048, fixed by the reference)

  float* out_newxyz = (float*)d_out;                                  // B*2048*3
  float* out_feat   = (float*)d_out + (size_t)BATCH * NPOINT * 3;     // B*128*2048
  int*   ws_idx     = (int*)d_ws;                                     // B*2048*32 ints = 2 MB

  ball_query_kernel<<<BATCH * 8, 256, 0, stream>>>(xyz, out_newxyz, ws_idx);
  grouped_mlp_kernel<<<256, 256, 0, stream>>>(xyz, feat, W1, b1, W2, b2,
                                              ws_idx, out_feat);
}
